// GNN_72121090835168
// MI455X (gfx1250) — matrix operand order repaired and checked by reading, not compiled, not run on hardware
//
#include <hip/hip_runtime.h>

// GNN message-passing on MI455X (gfx1250).
// f16 storage for edge states (halves the HBM traffic that dominates),
// f32 accumulation (WMMA C, segment-sum atomics). All GEMMs use
// v_wmma_f32_16x16x32_f16. Weights are pre-swizzled into per-lane
// fragment order so each B fragment is one contiguous 32B load.

#define NN    10000      // nodes
#define NE    160000     // edges (pairs 2k / 2k+1 are fwd/rev)
#define HID   300
#define NF    133
#define EF    14
#define NG    64         // graphs

#define HP    320        // hidden padded to 320 (20 tiles of 16, 10 K-chunks of 32)
#define NT    20         // column tiles
#define KC_H  10         // K chunks over HP

typedef __attribute__((ext_vector_type(16))) _Float16 v16h;
typedef __attribute__((ext_vector_type(8)))  float    v8f;

// A fragment (16x32 f16) from LDS, contiguous K order (this code's choice, not the ISA table order):
// lanes 0-15: row m=lane, halves 0..15 -> K k0..k0+15
// lanes 16-31: same row, K shifted by +16 (contiguous, matches B swizzle).
__device__ __forceinline__ v16h load_A_frag(const _Float16* As, int stride, int k0, int lane) {
  int m = lane & 15;
  int g = (lane >> 4) << 4;  // 0 or 16
  const _Float16* p = As + m * stride + k0 + g;
  v16h a;
#pragma unroll
  for (int i = 0; i < 8; ++i) a[i] = p[i];
#pragma unroll
  for (int i = 0; i < 8; ++i) a[8 + i] = p[8 + i];
  return a;
}

// ---------------- weight pre-swizzle --------------------------------------
// B fragment layout (32x16 f16): lane holds col n=lane&15, K = kc*32 + (lane>>4)*16 + i.
// Stored so fragment (kc,ct) for lane is 16 contiguous halves:
//   Wsw[(((kc*NT + ct)*32) + lane)*16 + i]
__global__ void k_swz(const float* __restrict__ W, _Float16* __restrict__ Wsw,
                      int KC, int Klim) {
  int t = blockIdx.x * blockDim.x + threadIdx.x;
  if (t >= KC * NT * 512) return;
  int i = t & 15, lane = (t >> 4) & 31;
  int rest = t >> 9;
  int ct = rest % NT, kc = rest / NT;
  int k = kc * 32 + ((lane >> 4) << 4) + i;
  int n = ct * 16 + (lane & 15);
  float v = (k < Klim && n < HID) ? W[(long)k * HID + n] : 0.f;
  Wsw[t] = (_Float16)v;
}

// W_en is (NF+HID, HID); padded K map: [0,160) <- rows [0,133), [160,480) <- rows [133,433)
__global__ void k_swz_en(const float* __restrict__ W, _Float16* __restrict__ Wsw) {
  const int KC = 15;
  int t = blockIdx.x * blockDim.x + threadIdx.x;
  if (t >= KC * NT * 512) return;
  int i = t & 15, lane = (t >> 4) & 31;
  int rest = t >> 9;
  int ct = rest % NT, kc = rest / NT;
  int k = kc * 32 + ((lane >> 4) << 4) + i;
  int n = ct * 16 + (lane & 15);
  float v = 0.f;
  if (n < HID) {
    if (k < NF)                       v = W[(long)k * HID + n];
    else if (k >= 160 && k < 160+HID) v = W[(long)(NF + k - 160) * HID + n];
  }
  Wsw[t] = (_Float16)v;
}

__global__ void k_padvec(const float* __restrict__ src, float* __restrict__ dst, int n) {
  int i = blockIdx.x * blockDim.x + threadIdx.x;
  if (i < HP) dst[i] = (i < n) ? src[i] : 0.f;
}

__global__ void k_zero(float* __restrict__ p, long n) {
  long i = (long)blockIdx.x * blockDim.x + threadIdx.x;
  if (i < n) p[i] = 0.f;
}

// ---------------- XW = x @ W_ei[:NF]  (per-node, K=133 pad 160) ------------
__global__ __launch_bounds__(128) void k_xw(const float* __restrict__ x,
    const _Float16* __restrict__ Wsw, float* __restrict__ XW) {
  const int KP = 160, AS = KP + 8;
  __shared__ _Float16 As[16 * AS];
  int n0 = blockIdx.x * 16;
  int tid = threadIdx.x;
  for (int idx = tid; idx < 16 * KP; idx += 128) {
    int j = idx / KP, k = idx - j * KP;
    float v = (k < NF) ? x[(long)(n0 + j) * NF + k] : 0.f;
    As[j * AS + k] = (_Float16)v;
  }
  __syncthreads();
  int lane = tid & 31, wave = tid >> 5;
  int cn = lane & 15, mo = (lane >> 4) * 8;
  for (int ct = wave; ct < NT; ct += 4) {
    v8f acc = {};
#pragma unroll
    for (int kc = 0; kc < 5; ++kc) {
      v16h a = load_A_frag(As, AS, kc * 32, lane);
      v16h b = *(const v16h*)(Wsw + (((long)kc * NT + ct) * 32 + lane) * 16);
      acc = __builtin_amdgcn_wmma_f32_16x16x32_f16(false, a, false, b, (short)0, acc, false, false);
    }
    int c = ct * 16 + cn;
#pragma unroll
    for (int r = 0; r < 8; ++r)
      XW[(long)(n0 + mo + r) * HP + c] = acc[r];
  }
}

// ---------------- h0 = relu(XW[row] + edge_attr@We + b); scatter a0 --------
__global__ __launch_bounds__(128) void k_h0(const float* __restrict__ ea,
    const _Float16* __restrict__ Wsw, const float* __restrict__ XW,
    const float* __restrict__ bias, const int* __restrict__ row,
    const int* __restrict__ colid, _Float16* __restrict__ h0,
    float* __restrict__ a0) {
  const int AS = 40;
  __shared__ _Float16 As[16 * AS];
  __shared__ int rtile[16], ctile[16];
  int e0 = blockIdx.x * 16;
  int tid = threadIdx.x;
  if (tid < 16) { rtile[tid] = row[e0 + tid]; ctile[tid] = colid[e0 + tid]; }
  for (int idx = tid; idx < 16 * 32; idx += 128) {
    int j = idx >> 5, k = idx & 31;
    float v = (k < EF) ? ea[(long)(e0 + j) * EF + k] : 0.f;
    As[j * AS + k] = (_Float16)v;
  }
  __syncthreads();
  int lane = tid & 31, wave = tid >> 5;
  int cn = lane & 15, mo = (lane >> 4) * 8;
  for (int ct = wave; ct < NT; ct += 4) {
    v16h a = load_A_frag(As, AS, 0, lane);
    v16h b = *(const v16h*)(Wsw + ((long)ct * 32 + lane) * 16);
    v8f acc = {};
    acc = __builtin_amdgcn_wmma_f32_16x16x32_f16(false, a, false, b, (short)0, acc, false, false);
    int c = ct * 16 + cn;
    float bv = bias[c];
#pragma unroll
    for (int r = 0; r < 8; ++r) {
      int j = mo + r;
      float v = acc[r] + bv + XW[(long)rtile[j] * HP + c];
      v = v > 0.f ? v : 0.f;
      h0[(long)(e0 + j) * HP + c] = (_Float16)v;
      atomicAdd(a0 + (long)ctile[j] * HP + c, v);
    }
  }
}

// ---------------- conv layer: h = relu((a[row]-rev(h))@Ws + bs + h0) -------
// Fuses next layer's segment_sum (scatter into anext). In-place safe: the
// 16-edge tile is pair-aligned, so rev (e^1) stays inside the tile; all reads
// of hprev happen before the barrier, writes after.
__global__ __launch_bounds__(128) void k_conv(
    const float* __restrict__ aprev, const _Float16* __restrict__ hprev,
    const _Float16* __restrict__ h0, const _Float16* __restrict__ Wsw,
    const float* __restrict__ bias, const int* __restrict__ row,
    const int* __restrict__ colid, _Float16* __restrict__ hout,
    float* __restrict__ anext) {
  const int AS = HP + 8;
  __shared__ _Float16 As[16 * AS];
  __shared__ int rtile[16], ctile[16];
  int e0 = blockIdx.x * 16;
  int tid = threadIdx.x;
  if (tid < 16) { rtile[tid] = row[e0 + tid]; ctile[tid] = colid[e0 + tid]; }
  __syncthreads();
  for (int idx = tid; idx < 16 * HP; idx += 128) {
    int j = idx / HP, k = idx - j * HP;
    float av = aprev[(long)rtile[j] * HP + k];
    float hv = (float)hprev[(long)((e0 + j) ^ 1) * HP + k];
    As[j * AS + k] = (_Float16)(av - hv);
  }
  __syncthreads();
  int lane = tid & 31, wave = tid >> 5;
  int cn = lane & 15, mo = (lane >> 4) * 8;
  for (int ct = wave; ct < NT; ct += 4) {
    v8f acc = {};
#pragma unroll
    for (int kc = 0; kc < KC_H; ++kc) {
      v16h a = load_A_frag(As, AS, kc * 32, lane);
      v16h b = *(const v16h*)(Wsw + (((long)kc * NT + ct) * 32 + lane) * 16);
      acc = __builtin_amdgcn_wmma_f32_16x16x32_f16(false, a, false, b, (short)0, acc, false, false);
    }
    int c = ct * 16 + cn;
    float bv = bias[c];
#pragma unroll
    for (int r = 0; r < 8; ++r) {
      int j = mo + r;
      float v = acc[r] + bv + (float)h0[(long)(e0 + j) * HP + c];
      v = v > 0.f ? v : 0.f;
      hout[(long)(e0 + j) * HP + c] = (_Float16)v;
      atomicAdd(anext + (long)ctile[j] * HP + c, v);
    }
  }
}

// ---------------- node MLP fused with graph pooling ------------------------
__global__ __launch_bounds__(128) void k_node(const float* __restrict__ x,
    const float* __restrict__ s, const _Float16* __restrict__ Wsw,
    const float* __restrict__ bias, const int* __restrict__ batch,
    float* __restrict__ g) {
  const int KP = 480, AS = 488;
  __shared__ _Float16 As[16 * AS];
  __shared__ int btile[16];
  int n0 = blockIdx.x * 16;
  int tid = threadIdx.x;
  if (tid < 16) btile[tid] = batch[n0 + tid];
  for (int idx = tid; idx < 16 * KP; idx += 128) {
    int j = idx / KP, k = idx - j * KP;
    float v = 0.f;
    if (k < NF)         v = x[(long)(n0 + j) * NF + k];
    else if (k >= 160)  v = s[(long)(n0 + j) * HP + (k - 160)];
    As[j * AS + k] = (_Float16)v;
  }
  __syncthreads();
  int lane = tid & 31, wave = tid >> 5;
  int cn = lane & 15, mo = (lane >> 4) * 8;
  for (int ct = wave; ct < NT; ct += 4) {
    v8f acc = {};
#pragma unroll
    for (int kc = 0; kc < 15; ++kc) {
      v16h a = load_A_frag(As, AS, kc * 32, lane);
      v16h b = *(const v16h*)(Wsw + (((long)kc * NT + ct) * 32 + lane) * 16);
      acc = __builtin_amdgcn_wmma_f32_16x16x32_f16(false, a, false, b, (short)0, acc, false, false);
    }
    int c = ct * 16 + cn;
    float bv = bias[c];
#pragma unroll
    for (int r = 0; r < 8; ++r) {
      int j = mo + r;
      float v = acc[r] + bv;
      v = v > 0.f ? v : 0.f;
      atomicAdd(g + (long)btile[j] * HP + c, v);
    }
  }
}

// ---------------- final readout: out = g @ W_ffn + b_ffn -------------------
__global__ void k_out(const float* __restrict__ g, const float* __restrict__ wf,
                      const float* __restrict__ bf, float* __restrict__ out) {
  int gi = threadIdx.x;
  if (gi < NG) {
    float s = bf[0];
    for (int k = 0; k < HID; ++k) s += g[gi * HP + k] * wf[k];
    out[gi] = s;
  }
}

// ---------------------------------------------------------------------------
extern "C" void kernel_launch(void* const* d_in, const int* in_sizes, int n_in,
                              void* d_out, int out_size, void* d_ws, size_t ws_size,
                              hipStream_t stream) {
  (void)in_sizes; (void)n_in; (void)out_size; (void)ws_size;
  const float* x    = (const float*)d_in[0];
  const int*   row  = (const int*)d_in[1];          // edge_index[0]
  const int*   col  = row + NE;                     // edge_index[1]
  const float* ea   = (const float*)d_in[2];
  const int*   bat  = (const int*)d_in[3];
  const float* W_ei = (const float*)d_in[4];
  const float* b_ei = (const float*)d_in[5];
  const float* Ws   = (const float*)d_in[6];
  const float* bs   = (const float*)d_in[7];
  const float* W_en = (const float*)d_in[8];
  const float* b_en = (const float*)d_in[9];
  const float* Wffn = (const float*)d_in[10];
  const float* bffn = (const float*)d_in[11];
  float* out = (float*)d_out;

  char* p = (char*)d_ws;
  float*    XW     = (float*)p;    p += (long)NN * HP * 4;
  _Float16* h0     = (_Float16*)p; p += (long)NE * HP * 2;
  _Float16* h      = (_Float16*)p; p += (long)NE * HP * 2;
  float*    aA     = (float*)p;    p += (long)NN * HP * 4;
  float*    aB     = (float*)p;    p += (long)NN * HP * 4;
  float*    g      = (float*)p;    p += (long)NG * HP * 4;
  _Float16* Wx_sw  = (_Float16*)p; p += 5  * NT * 512 * 2;
  _Float16* We_sw  = (_Float16*)p; p += 1  * NT * 512 * 2;
  _Float16* Ws_sw  = (_Float16*)p; p += 3L * KC_H * NT * 512 * 2;
  _Float16* Wen_sw = (_Float16*)p; p += 15 * NT * 512 * 2;
  float*    bei    = (float*)p;    p += HP * 4;
  float*    bsp    = (float*)p;    p += 3 * HP * 4;
  float*    ben    = (float*)p;    p += HP * 4;

  // weight conversion / swizzle
  k_swz<<<(5*NT*512+255)/256, 256, 0, stream>>>(W_ei, Wx_sw, 5, NF);
  k_swz<<<(1*NT*512+255)/256, 256, 0, stream>>>(W_ei + (long)NF*HID, We_sw, 1, EF);
  for (int l = 0; l < 3; ++l)
    k_swz<<<(KC_H*NT*512+255)/256, 256, 0, stream>>>(Ws + (long)l*HID*HID,
        Ws_sw + (long)l*KC_H*NT*512, KC_H, HID);
  k_swz_en<<<(15*NT*512+255)/256, 256, 0, stream>>>(W_en, Wen_sw);
  k_padvec<<<2, 256, 0, stream>>>(b_ei, bei, HID);
  for (int l = 0; l < 3; ++l)
    k_padvec<<<2, 256, 0, stream>>>(bs + (long)l*HID, bsp + l*HP, HID);
  k_padvec<<<2, 256, 0, stream>>>(b_en, ben, HID);

  long an = (long)NN * HP;
  k_zero<<<(int)((an+255)/256), 256, 0, stream>>>(aA, an);
  k_zero<<<(int)((NG*HP+255)/256), 256, 0, stream>>>(g, NG*HP);

  // node-feature projection + edge init (scatters a0 into aA)
  k_xw<<<NN/16, 128, 0, stream>>>(x, Wx_sw, XW);
  k_h0<<<NE/16, 128, 0, stream>>>(ea, We_sw, XW, bei, row, col, h0, aA);

  // three conv layers, ping-pong aggregation buffers, fused scatter
  k_zero<<<(int)((an+255)/256), 256, 0, stream>>>(aB, an);
  k_conv<<<NE/16, 128, 0, stream>>>(aA, h0, h0, Ws_sw,               bsp,        row, col, h, aB);
  k_zero<<<(int)((an+255)/256), 256, 0, stream>>>(aA, an);
  k_conv<<<NE/16, 128, 0, stream>>>(aB, h,  h0, Ws_sw + KC_H*NT*512,   bsp + HP,   row, col, h, aA);
  k_zero<<<(int)((an+255)/256), 256, 0, stream>>>(aB, an);
  k_conv<<<NE/16, 128, 0, stream>>>(aA, h,  h0, Ws_sw + 2L*KC_H*NT*512, bsp + 2*HP, row, col, h, aB);

  // node MLP + graph pooling + readout
  k_node<<<NN/16, 128, 0, stream>>>(x, aB, Wen_sw, ben, bat, g);
  k_out<<<1, 64, 0, stream>>>(g, Wffn, bffn, out);
}
